// MultiHeadSectionAttentionImputer_29764123361477
// MI455X (gfx1250) — matrix operand order repaired and checked by reading, not compiled, not run on hardware
//
#include <hip/hip_runtime.h>
#include <stdint.h>

#define NSEC   8192
#define EDIM   768
#define NHEADS 12
#define DK     64
#define TDIM   768
#define MMISS  2048
#define NEXIST 6144

typedef __attribute__((ext_vector_type(16))) __bf16 v16bf;
typedef __attribute__((ext_vector_type(8)))  float  v8f;
typedef unsigned short u16;

// ---------------- helpers ----------------

static __device__ __forceinline__ u16 f2bf(float f) {
    union { float f; unsigned u; } x; x.f = f;
    unsigned u = x.u;
    u += 0x7FFFu + ((u >> 16) & 1u);   // round-to-nearest-even
    return (u16)(u >> 16);
}

// Build a v16bf A/B fragment from two 16-byte chunks (8 bf16 each). Works for
// global or LDS pointers (addrspace inferred -> global_load_b128 / ds_load_b128).
static __device__ __forceinline__ v16bf ld_frag(const u16* p0, const u16* p1) {
    union { v16bf v; uint4 q[2]; } u;
    u.q[0] = *(const uint4*)p0;
    u.q[1] = *(const uint4*)p1;
    return u.v;
}

static __device__ __forceinline__ v8f wmma_bf16(v16bf a, v16bf b, v8f c) {
    return __builtin_amdgcn_wmma_f32_16x16x32_bf16(
        /*neg_a=*/false, a, /*neg_b=*/false, b,
        /*c_mod=*/(short)0, c, /*reuse_a=*/false, /*reuse_b=*/false);
}

// Machine-scheduler fence: keep the fragment-load group above the WMMA group
// so DS loads issue together and WMMAs run back-to-back.
static __device__ __forceinline__ void sched_fence() {
#if defined(__has_builtin)
#if __has_builtin(__builtin_amdgcn_sched_barrier)
    __builtin_amdgcn_sched_barrier(0);
#endif
#endif
}

// CDNA5 async global->LDS copy (tracked by ASYNCcnt). LDS byte address is the
// low 32 bits of the generic pointer.
static __device__ __forceinline__ void async_ld_b128(const u16* gptr, u16* lptr) {
    unsigned lds = (unsigned)(uintptr_t)lptr;
    unsigned long long ga = (unsigned long long)(uintptr_t)gptr;
    asm volatile("global_load_async_to_lds_b128 %0, %1, off"
                 :: "v"(lds), "v"(ga) : "memory");
}
static __device__ __forceinline__ void wait_asynccnt0() {
    asm volatile("s_wait_asynccnt 0x0" ::: "memory");
}

// ---------------- prep kernels ----------------

__global__ void copy_f32_kernel(const float* __restrict__ src, float* __restrict__ dst, int n) {
    int i = blockIdx.x * blockDim.x + threadIdx.x;
    if (i < n) dst[i] = src[i];
}

// Wt[o][e] = bf16(W[e][o])   (transpose so B-fragment K runs contiguously)
__global__ void cvt_w_t_kernel(const float* __restrict__ W, u16* __restrict__ Wt) {
    int i = blockIdx.x * blockDim.x + threadIdx.x;
    if (i >= EDIM * TDIM) return;
    int o = i / EDIM, e = i % EDIM;
    Wt[i] = f2bf(W[(size_t)e * TDIM + o]);
}

// out[r][e] = bf16(ehr[idx[r]][e])
__global__ void gather_rows_kernel(const float* __restrict__ ehr, const int* __restrict__ idx,
                                   u16* __restrict__ out, int rows) {
    int i = blockIdx.x * blockDim.x + threadIdx.x;
    if (i >= rows * EDIM) return;
    int r = i / EDIM, e = i % EDIM;
    out[i] = f2bf(ehr[(size_t)idx[r] * EDIM + e]);
}

// out[(h*rows + r)*64 + d] = bf16(cooc[(h*8192 + idx[r])*64 + d])
__global__ void gather_bias_kernel(const float* __restrict__ cooc, const int* __restrict__ idx,
                                   u16* __restrict__ out, int rows) {
    int i = blockIdx.x * blockDim.x + threadIdx.x;
    if (i >= NHEADS * rows * DK) return;
    int d = i % DK; int t = i / DK; int r = t % rows; int h = t / rows;
    out[i] = f2bf(cooc[((size_t)h * NSEC + idx[r]) * DK + d]);
}

// ---------------- WMMA GEMM (LDS-staged, async double-buffered) ----------------
// One block = 128 rows x 64 output cols (8 waves x 16 rows).  Per K-step (32),
// the block async-stages A (128x32) and W (64x32) tiles into LDS; all waves
// consume via ds_load_b128.  Rows padded to 40 ush (20 dwords) for bank spread.

struct __align__(16) GemmLds {
    u16 at[2][128][40];   // 20480 B
    u16 bt[2][64][40];    // 10240 B
};

__global__ void __launch_bounds__(256)
gemm_bf16_wmma(const u16* __restrict__ A, const u16* __restrict__ Wt,
               const float* __restrict__ bias, u16* __restrict__ out,
               int rowsA, float scale, int transposed)
{
    __shared__ GemmLds S;
    const int tid  = threadIdx.x;
    const int wib  = tid >> 5;
    const int lane = tid & 31;
    const int half = lane >> 4, lq = lane & 15;

    const int ot    = blockIdx.x % (TDIM / 64);   // 12 tile columns
    const int mgrp  = blockIdx.x / (TDIM / 64);
    const int obase = ot * 64;
    const int mblk  = mgrp * 128;
    const int mbase = mblk + wib * 16;

    auto stage = [&](int k, int buf) {
        // A tile: 128 rows x 4 b128 chunks = 512, 2 per thread
        int id = tid;
        int r = id >> 2, c = id & 3;
        async_ld_b128(A + (size_t)(mblk + r) * EDIM + k + c * 8, &S.at[buf][r][c * 8]);
        id = tid + 256; r = id >> 2; c = id & 3;
        async_ld_b128(A + (size_t)(mblk + r) * EDIM + k + c * 8, &S.at[buf][r][c * 8]);
        // W tile: 64 rows x 4 chunks = 256, 1 per thread
        r = tid >> 2; c = tid & 3;
        async_ld_b128(Wt + (size_t)(obase + r) * EDIM + k + c * 8, &S.bt[buf][r][c * 8]);
    };

    v8f zero = {};
    v8f c[4] = { zero, zero, zero, zero };

    int buf = 0;
    stage(0, 0);
    wait_asynccnt0();
    __syncthreads();

    for (int k = 0; k < EDIM; k += 32) {
        int kn = k + 32;
        if (kn < EDIM) stage(kn, buf ^ 1);          // overlap with compute

        // all fragment loads issue first (sched fence), WMMAs then run
        // back-to-back with decreasing dscnt waits
        const u16* ar = &S.at[buf][wib * 16 + lq][0];
        const u16* b0 = &S.bt[buf][ 0 + lq][half * 16];
        const u16* b1 = &S.bt[buf][16 + lq][half * 16];
        const u16* b2 = &S.bt[buf][32 + lq][half * 16];
        const u16* b3 = &S.bt[buf][48 + lq][half * 16];
        v16bf af  = ld_frag(ar + half * 16, ar + half * 16 + 8);
        v16bf bf0 = ld_frag(b0, b0 + 8);
        v16bf bf1 = ld_frag(b1, b1 + 8);
        v16bf bf2 = ld_frag(b2, b2 + 8);
        v16bf bf3 = ld_frag(b3, b3 + 8);
        sched_fence();
        c[0] = wmma_bf16(af, bf0, c[0]);
        c[1] = wmma_bf16(af, bf1, c[1]);
        c[2] = wmma_bf16(af, bf2, c[2]);
        c[3] = wmma_bf16(af, bf3, c[3]);

        wait_asynccnt0();
        __syncthreads();
        buf ^= 1;
    }

#pragma unroll
    for (int t = 0; t < 4; ++t) {
        int col = obase + t * 16 + lq;
        float bb = bias[col];
#pragma unroll
        for (int r = 0; r < 8; ++r) {
            int row = mbase + r + 8 * half;
            u16 v = f2bf((c[t][r] + bb) * scale);
            if (transposed) out[(size_t)col * rowsA + row] = v;   // vT[o][n]
            else            out[(size_t)row * TDIM + col] = v;    // q/k[m][o]
        }
    }
}

// ---------------- flash attention ----------------
// One block = one head x 128 query rows (8 waves x 16 rows). The block
// double-buffers K / exist-bias / V^T chunks (32 exist columns) in LDS via
// async global->LDS copies; all 8 waves consume the shared tiles.
// Rows padded (72 ush = 36 dwords, 40 ush = 20 dwords) for bank spread.

struct __align__(16) AttnLds {
    u16 kt[2][32][72];    // K rows       9216 B
    u16 et[2][32][72];    // exist bias   9216 B
    u16 vt[2][64][40];    // V^T [d][n]  10240 B
    u16 p[8][16][40];     // P tiles     10240 B
};

__global__ void __launch_bounds__(256)
attn_wmma(const u16* __restrict__ qbf, const u16* __restrict__ kbf,
          const u16* __restrict__ vT,  const u16* __restrict__ mb,
          const u16* __restrict__ eb,  const int* __restrict__ missing_idx,
          float* __restrict__ out)
{
    __shared__ AttnLds S;

    const int tid  = threadIdx.x;
    const int wib  = tid >> 5;
    const int lane = tid & 31;
    const int half = lane >> 4, lq = lane & 15;

    const int h     = blockIdx.x >> 4;                  // 12 heads
    const int mgrp  = blockIdx.x & 15;                  // 16 groups of 128 rows
    const int mbase = mgrp * 128 + wib * 16;

    // Q (1/8 scale folded in at projection) and missing-bias A fragments (K = d)
    const u16* qrow = qbf + (size_t)(mbase + lq) * TDIM + h * DK;
    const u16* mrow = mb  + ((size_t)h * MMISS + mbase + lq) * DK;
    v16bf aq0 = ld_frag(qrow + half * 16,      qrow + half * 16 + 8);
    v16bf aq1 = ld_frag(qrow + 32 + half * 16, qrow + 32 + half * 16 + 8);
    v16bf am0 = ld_frag(mrow + half * 16,      mrow + half * 16 + 8);
    v16bf am1 = ld_frag(mrow + 32 + half * 16, mrow + 32 + half * 16 + 8);

    v8f zero = {};
    v8f o0 = zero, o1 = zero, o2 = zero, o3 = zero;
    float m_run[8], l_run[8];
#pragma unroll
    for (int r = 0; r < 8; ++r) { m_run[r] = -3.0e38f; l_run[r] = 0.0f; }

    // cooperative async stage of one 32-column chunk into LDS buffer `buf`
    auto stage_chunk = [&](int nb, int buf) {
        int r = tid >> 3, c = tid & 7;   // 32 rows x 8 chunks
        async_ld_b128(kbf + (size_t)(nb + r) * TDIM + h * DK + c * 8,
                      &S.kt[buf][r][c * 8]);
        async_ld_b128(eb + ((size_t)h * NEXIST + nb + r) * DK + c * 8,
                      &S.et[buf][r][c * 8]);
        int d = tid >> 2, c2 = tid & 3;  // 64 d-rows x 4 chunks
        async_ld_b128(vT + (size_t)(h * DK + d) * NEXIST + nb + c2 * 8,
                      &S.vt[buf][d][c2 * 8]);
    };

    int buf = 0;
    stage_chunk(0, 0);
    wait_asynccnt0();
    __syncthreads();

    for (int nb = 0; nb < NEXIST; nb += 32) {
        int nxt = nb + 32;
        if (nxt < NEXIST) stage_chunk(nxt, buf ^ 1);    // overlap with compute

        // ---- all fragment loads for this chunk issue first ----
        const u16* k0 = &S.kt[buf][lq][0];
        const u16* k1 = &S.kt[buf][16 + lq][0];
        const u16* e0 = &S.et[buf][lq][0];
        const u16* e1 = &S.et[buf][16 + lq][0];
        v16bf bk00 = ld_frag(k0 + half * 16,      k0 + half * 16 + 8);
        v16bf bk01 = ld_frag(k0 + 32 + half * 16, k0 + 32 + half * 16 + 8);
        v16bf be00 = ld_frag(e0 + half * 16,      e0 + half * 16 + 8);
        v16bf be01 = ld_frag(e0 + 32 + half * 16, e0 + 32 + half * 16 + 8);
        v16bf bk10 = ld_frag(k1 + half * 16,      k1 + half * 16 + 8);
        v16bf bk11 = ld_frag(k1 + 32 + half * 16, k1 + 32 + half * 16 + 8);
        v16bf be10 = ld_frag(e1 + half * 16,      e1 + half * 16 + 8);
        v16bf be11 = ld_frag(e1 + 32 + half * 16, e1 + 32 + half * 16 + 8);
        const u16* v0 = &S.vt[buf][ 0 + lq][half * 16];
        const u16* v1 = &S.vt[buf][16 + lq][half * 16];
        const u16* v2 = &S.vt[buf][32 + lq][half * 16];
        const u16* v3 = &S.vt[buf][48 + lq][half * 16];
        v16bf bv0 = ld_frag(v0, v0 + 8);
        v16bf bv1 = ld_frag(v1, v1 + 8);
        v16bf bv2 = ld_frag(v2, v2 + 8);
        v16bf bv3 = ld_frag(v3, v3 + 8);
        sched_fence();

        // ---- 8 score WMMAs back-to-back ----
        v8f s0 = zero, s1 = zero;
        s0 = wmma_bf16(aq0, bk00, s0);
        s0 = wmma_bf16(aq1, bk01, s0);
        s0 = wmma_bf16(am0, be00, s0);
        s0 = wmma_bf16(am1, be01, s0);
        s1 = wmma_bf16(aq0, bk10, s1);
        s1 = wmma_bf16(aq1, bk11, s1);
        s1 = wmma_bf16(am0, be10, s1);
        s1 = wmma_bf16(am1, be11, s1);

        // online softmax over 32 columns; row r lives in one 16-lane half
#pragma unroll
        for (int r = 0; r < 8; ++r) {
            float cm = fmaxf(s0[r], s1[r]);
            cm = fmaxf(cm, __shfl_xor(cm, 1, 32));
            cm = fmaxf(cm, __shfl_xor(cm, 2, 32));
            cm = fmaxf(cm, __shfl_xor(cm, 4, 32));
            cm = fmaxf(cm, __shfl_xor(cm, 8, 32));
            float mn = fmaxf(m_run[r], cm);
            float a  = __expf(m_run[r] - mn);
            float p0 = __expf(s0[r] - mn);
            float p1 = __expf(s1[r] - mn);
            s0[r] = p0; s1[r] = p1;
            float rs = p0 + p1;
            rs += __shfl_xor(rs, 1, 32);
            rs += __shfl_xor(rs, 2, 32);
            rs += __shfl_xor(rs, 4, 32);
            rs += __shfl_xor(rs, 8, 32);
            l_run[r] = l_run[r] * a + rs;
            m_run[r] = mn;
            o0[r] *= a; o1[r] *= a; o2[r] *= a; o3[r] *= a;
        }

        // re-layout P (C-layout f32) -> A fragment via per-wave LDS tile
        // (DS ops are in-order within a wave; no barrier needed)
#pragma unroll
        for (int r = 0; r < 8; ++r) {
            int row = r + 8 * half;
            S.p[wib][row][lq]      = f2bf(s0[r]);
            S.p[wib][row][16 + lq] = f2bf(s1[r]);
        }
        const u16* prow = &S.p[wib][lq][0];
        v16bf ap = ld_frag(prow + half * 16, prow + half * 16 + 8);

        // ---- 4 PV WMMAs back-to-back on pre-loaded V fragments ----
        o0 = wmma_bf16(ap, bv0, o0);
        o1 = wmma_bf16(ap, bv1, o1);
        o2 = wmma_bf16(ap, bv2, o2);
        o3 = wmma_bf16(ap, bv3, o3);

        wait_asynccnt0();     // this wave's staging issues complete
        __syncthreads();      // whole block: next buffer ready, current consumed
        buf ^= 1;
    }

    // normalize and scatter into output rows
#pragma unroll
    for (int r = 0; r < 8; ++r) {
        float inv = 1.0f / l_run[r];
        int row = mbase + r + 8 * half;
        int sec = missing_idx[row];
        float* orow = out + (size_t)sec * EDIM + h * DK;
        orow[ 0 + lq] = o0[r] * inv;
        orow[16 + lq] = o1[r] * inv;
        orow[32 + lq] = o2[r] * inv;
        orow[48 + lq] = o3[r] * inv;
    }
}

// ---------------- launcher ----------------

extern "C" void kernel_launch(void* const* d_in, const int* in_sizes, int n_in,
                              void* d_out, int out_size, void* d_ws, size_t ws_size,
                              hipStream_t stream) {
    const float* ehr   = (const float*)d_in[0];
    const int* exist_i = (const int*)d_in[1];
    const int* miss_i  = (const int*)d_in[2];
    const float* Wq    = (const float*)d_in[3];
    const float* bq    = (const float*)d_in[4];
    const float* Wk    = (const float*)d_in[5];
    const float* bk    = (const float*)d_in[6];
    const float* Wv    = (const float*)d_in[7];
    const float* bv    = (const float*)d_in[8];
    const float* cooc  = (const float*)d_in[9];
    float* out = (float*)d_out;

    char* ws = (char*)d_ws;
    size_t off = 0;
    auto carve = [&](size_t bytes) -> void* {
        void* p = ws + off;
        off += (bytes + 255) & ~(size_t)255;
        return p;
    };
    u16* q_in  = (u16*)carve((size_t)MMISS  * EDIM * 2);
    u16* kv_in = (u16*)carve((size_t)NEXIST * EDIM * 2);
    u16* Wt_q  = (u16*)carve((size_t)EDIM * TDIM * 2);
    u16* Wt_k  = (u16*)carve((size_t)EDIM * TDIM * 2);
    u16* Wt_v  = (u16*)carve((size_t)EDIM * TDIM * 2);
    u16* q_bf  = (u16*)carve((size_t)MMISS  * TDIM * 2);
    u16* k_bf  = (u16*)carve((size_t)NEXIST * TDIM * 2);
    u16* vT    = (u16*)carve((size_t)TDIM * NEXIST * 2);
    u16* mb    = (u16*)carve((size_t)NHEADS * MMISS  * DK * 2);
    u16* eb    = (u16*)carve((size_t)NHEADS * NEXIST * DK * 2);

    const int TB = 256;
    auto blocks = [](long n, int tb) { return (int)((n + tb - 1) / tb); };

    // 1) base copy of table into output
    copy_f32_kernel<<<blocks((long)NSEC * EDIM, TB), TB, 0, stream>>>(ehr, out, NSEC * EDIM);

    // 2) convert + transpose weights
    cvt_w_t_kernel<<<blocks((long)EDIM * TDIM, TB), TB, 0, stream>>>(Wq, Wt_q);
    cvt_w_t_kernel<<<blocks((long)EDIM * TDIM, TB), TB, 0, stream>>>(Wk, Wt_k);
    cvt_w_t_kernel<<<blocks((long)EDIM * TDIM, TB), TB, 0, stream>>>(Wv, Wt_v);

    // 3) gather + convert embeddings and biases
    gather_rows_kernel<<<blocks((long)MMISS * EDIM, TB), TB, 0, stream>>>(ehr, miss_i, q_in, MMISS);
    gather_rows_kernel<<<blocks((long)NEXIST * EDIM, TB), TB, 0, stream>>>(ehr, exist_i, kv_in, NEXIST);
    gather_bias_kernel<<<blocks((long)NHEADS * MMISS * DK, TB), TB, 0, stream>>>(cooc, miss_i, mb, MMISS);
    gather_bias_kernel<<<blocks((long)NHEADS * NEXIST * DK, TB), TB, 0, stream>>>(cooc, exist_i, eb, NEXIST);

    // 4) WMMA projections (LDS-staged).  Q gets the 1/sqrt(64) fold; V transposed.
    {
        int blocksQ  = (MMISS  / 128) * (TDIM / 64);   // 192
        int blocksKV = (NEXIST / 128) * (TDIM / 64);   // 576
        gemm_bf16_wmma<<<blocksQ,  TB, 0, stream>>>(q_in,  Wt_q, bq, q_bf, MMISS, 0.125f, 0);
        gemm_bf16_wmma<<<blocksKV, TB, 0, stream>>>(kv_in, Wt_k, bk, k_bf, NEXIST, 1.0f, 0);
        gemm_bf16_wmma<<<blocksKV, TB, 0, stream>>>(kv_in, Wt_v, bv, vT,   NEXIST, 1.0f, 1);
    }

    // 5) fused flash attention + scatter (12 heads x 16 row-groups)
    attn_wmma<<<NHEADS * 16, TB, 0, stream>>>(q_bf, k_bf, vT, mb, eb, miss_i, out);
}
